// MultiheadAttention_55344948576836
// MI455X (gfx1250) — matrix operand order repaired and checked by reading, not compiled, not run on hardware
//
#include <hip/hip_runtime.h>
#include <math.h>

// ---------------------------------------------------------------------------
// TransformerXL-style relative MHA for MI455X (gfx1250), wave32 + WMMA bf16.
// - NT GEMMs (pre-transposed bf16 weights) so every WMMA fragment is two
//   16-byte lane-contiguous loads.
// - A panels staged block-wide through LDS with global_load_async_to_lds_b128
//   (ASYNCcnt) since all 8 waves of a block share the same 32 A rows.
// - Fused flash attention with the relative-shift band trick.
// ---------------------------------------------------------------------------

typedef __bf16 bf16_t;
typedef bf16_t v16bf __attribute__((ext_vector_type(16)));
typedef bf16_t v8bf  __attribute__((ext_vector_type(8)));
typedef float  v8f   __attribute__((ext_vector_type(8)));
typedef float  v4f   __attribute__((ext_vector_type(4)));

constexpr int BB   = 2;
constexpr int T    = 2048;
constexpr int H    = 8;
constexpr int CK   = 64;
constexpr int CV   = 192;
constexpr int EMB  = H * CV;    // 1536
constexpr int KPROJ= H * CK;    // 512
constexpr int NPF  = 192;
constexpr int LPAD = 4096;      // padded 2T-1 = 4095 -> 4096 rows
constexpr int MROWS= BB * T;    // 4096

__device__ inline v8f wmma_bf16(v16bf a, v16bf b, v8f c) {
  return __builtin_amdgcn_wmma_f32_16x16x32_bf16(
      false, a, false, b, (short)0, c, false, false);
}

// Fragment from two contiguous 8-element (16B) chunks.
__device__ inline v16bf ld_frag(const bf16_t* p0, const bf16_t* p1) {
  union { v16bf f; v8bf h[2]; } u;
  u.h[0] = *reinterpret_cast<const v8bf*>(p0);
  u.h[1] = *reinterpret_cast<const v8bf*>(p1);
  return u.f;
}

// ---------------------------------------------------------------------------
// NT GEMM: C[M,N] = A[M,K](bf16 row-major) @ BT[N,K](bf16 row-major)^T
// One wave computes a 32x64 C macro-tile (8 accumulators, 8 WMMAs / K-step).
// All 8 waves of a block share tm, so the 32x64 A panel is staged once per
// block via async global->LDS (one b128 per thread), then consumed with
// ds_load_b128 fragment reads. B fragments stream straight from global.
// ---------------------------------------------------------------------------
template <bool OUTF32, bool BIAS, bool TRANSB16>
__global__ __launch_bounds__(256)
void gemm_nt(const bf16_t* __restrict__ A, const bf16_t* __restrict__ BT,
             float* __restrict__ Cf, bf16_t* __restrict__ Cb,
             const float* __restrict__ bias, int M, int N, int K) {
  __shared__ alignas(16) bf16_t slabA[32 * 64];   // 4KB A panel (32 rows x 64 k)

  const int tid  = threadIdx.x;
  const int lane = tid & 31;
  const int row  = lane & 15;
  const int hi   = lane >> 4;
  const int wid  = blockIdx.x * 8 + (tid >> 5);
  const int ncol = N >> 6;              // 64-wide col tiles
  const int tm   = wid / ncol;          // 32-row tile (uniform across block)
  const int tn   = wid % ncol;

  // async A-panel staging: thread t loads 8 elems of row t/8, chunk (t%8)*8
  const int lrow = tid >> 3;
  const int lcol = (tid & 7) * 8;
  const unsigned lds_dst =
      (unsigned)(size_t)&slabA[lrow * 64 + lcol];       // raw LDS byte offset
  const bf16_t* gsrc_base = A + (size_t)(tm * 32 + lrow) * K + lcol;

  const bf16_t* Brow[4];
#pragma unroll
  for (int c = 0; c < 4; ++c)
    Brow[c] = BT + (size_t)(tn * 64 + c * 16 + row) * K;

  v8f acc[2][4];
#pragma unroll
  for (int rt = 0; rt < 2; ++rt)
#pragma unroll
    for (int c = 0; c < 4; ++c) acc[rt][c] = v8f{};

  const bf16_t* a0base = &slabA[row * 64];
  const bf16_t* a1base = &slabA[(row + 16) * 64];

  for (int kp = 0; kp < K; kp += 64) {
    const bf16_t* gsrc = gsrc_base + kp;
    asm volatile("global_load_async_to_lds_b128 %0, %1, off"
                 :: "v"(lds_dst), "v"(gsrc) : "memory");
    asm volatile("s_wait_asynccnt 0x0" ::: "memory");
    __syncthreads();
#pragma unroll
    for (int ks = 0; ks < 64; ks += 32) {
      const int ka = ks + hi * 8;               // A chunks: ka, ka+16 (in LDS)
      const int kb = kp + ks + hi * 8;          // B chunks: kb, kb+16 (global)
      const v16bf a0 = ld_frag(a0base + ka, a0base + ka + 16);
      const v16bf a1 = ld_frag(a1base + ka, a1base + ka + 16);
#pragma unroll
      for (int c = 0; c < 4; ++c) {
        const v16bf b = ld_frag(Brow[c] + kb, Brow[c] + kb + 16);
        acc[0][c] = wmma_bf16(a0, b, acc[0][c]);
        acc[1][c] = wmma_bf16(a1, b, acc[1][c]);
      }
    }
    __syncthreads();
  }

#pragma unroll
  for (int rt = 0; rt < 2; ++rt)
#pragma unroll
    for (int c = 0; c < 4; ++c) {
      const int colg = tn * 64 + c * 16 + row;
      const int rbase = tm * 32 + rt * 16 + hi * 8;
      if (TRANSB16) {
        v8bf pk;
#pragma unroll
        for (int r = 0; r < 8; ++r) pk[r] = (bf16_t)acc[rt][c][r];
        *reinterpret_cast<v8bf*>(Cb + (size_t)colg * M + rbase) = pk;
      } else {
#pragma unroll
        for (int r = 0; r < 8; ++r) {
          float v = acc[rt][c][r];
          if (BIAS) v += bias[colg];
          if (OUTF32) Cf[(size_t)(rbase + r) * N + colg] = v;
          else        Cb[(size_t)(rbase + r) * N + colg] = (bf16_t)v;
        }
      }
    }
}

// ---------------------------------------------------------------------------
// Positional central-mask features -> bf16 (LPAD x NPF), rows >= 2T-1 zeroed.
// ---------------------------------------------------------------------------
__global__ void pos_enc_kernel(bf16_t* __restrict__ pe) {
  const int l = blockIdx.x;
  const int i = threadIdx.x;
  float out = 0.f;
  if (l < 2 * T - 1) {
    const float d = (float)l - (float)(T - 1);
    const int nb = NPF / 2;
    const int j = i % nb;
    const float log_rate = __logf((float)T + 1.f) / (float)nb;
    const float cw = __expf((float)(j + 1) * log_rate) - 1.f;
    const float emb = (cw > fabsf(d)) ? 1.f : 0.f;
    const float sgn = (d > 0.f) ? 1.f : ((d < 0.f) ? -1.f : 0.f);
    out = (i < nb) ? emb : sgn * emb;
  }
  pe[(size_t)l * NPF + i] = (bf16_t)out;
}

__global__ void cast_f32_bf16(const float* __restrict__ in,
                              bf16_t* __restrict__ out, int n) {
  const int i = blockIdx.x * 256 + threadIdx.x;
  if (i < n) out[i] = (bf16_t)in[i];
}

// in (R,C) f32 row-major -> out (C,R) bf16 row-major (weights: cast once).
__global__ void cast_transpose_f32_bf16(const float* __restrict__ in,
                                        bf16_t* __restrict__ out,
                                        int R, int C) {
  const int c = blockIdx.x * 32 + (threadIdx.x & 31);
  const int r = blockIdx.y * 8 + (threadIdx.x >> 5);
  if (c < C && r < R) out[(size_t)c * R + r] = (bf16_t)in[(size_t)r * C + c];
}

// ---------------------------------------------------------------------------
// Fused flash attention: one wave = 16 queries of one (b,h); online softmax
// over 32-key tiles. Relative logits via a 31-row r_k band + LDS diagonal
// gather (rel[q,j] = qr . r_k[j - q + T - 1]).  LDS scratch is wave-private
// and CDNA5 LDS ops are in-order within a wave, so no workgroup barriers.
// ---------------------------------------------------------------------------
__global__ __launch_bounds__(128)
void attn_kernel(const float* __restrict__ qf,   // (B,T,KPROJ) f32
                 const bf16_t* __restrict__ kb,  // (B,T,KPROJ) bf16
                 const bf16_t* __restrict__ vt,  // (EMB, B*T) bf16 (transposed)
                 const bf16_t* __restrict__ rkb, // (LPAD,KPROJ) bf16
                 const float* __restrict__ rwb,  // (H*CK)
                 const float* __restrict__ rrb,  // (H*CK)
                 bf16_t* __restrict__ ob) {      // (B,T,EMB) bf16
  __shared__ alignas(16) float lds[4][16 * 32];

  const int lane = threadIdx.x & 31;
  const int wv   = threadIdx.x >> 5;
  const int row  = lane & 15;
  const int hi   = lane >> 4;
  float* P = lds[wv];

  const int QT = T / 16;
  const int w  = blockIdx.x * 4 + wv;
  const int qt = w % QT;
  const int h  = (w / QT) % H;
  const int b  = w / (QT * H);
  const int q0 = qt * 16;

  // Q fragments with both biases, pre-scaled by 1/sqrt(CK)
  v16bf aqw[2], aqr[2];
  const float* Qrow = qf + (size_t)(b * T + q0 + row) * KPROJ + h * CK;
#pragma unroll
  for (int ch = 0; ch < 2; ++ch) {
    const int kb0 = ch * 32 + hi * 16;
#pragma unroll
    for (int i = 0; i < 16; ++i) {
      const int kk = kb0 + i;
      const float qv = Qrow[kk] * 0.125f;
      aqw[ch][i] = (bf16_t)(qv + rwb[h * CK + kk]);
      aqr[ch][i] = (bf16_t)(qv + rrb[h * CK + kk]);
    }
  }

  v8f mI, lI, O[12];
#pragma unroll
  for (int r = 0; r < 8; ++r) { mI[r] = -3.0e38f; lI[r] = 0.f; }
#pragma unroll
  for (int t = 0; t < 12; ++t) O[t] = v8f{};

  for (int j0 = 0; j0 < T; j0 += 32) {
    v8f S[2];
#pragma unroll
    for (int cl = 0; cl < 2; ++cl) {
      const int jc = j0 + cl * 16;
      // content logits: (q + r_w_bias) . k   (K fragment is lane-contiguous)
      v8f acc = {};
#pragma unroll
      for (int ch = 0; ch < 2; ++ch) {
        const bf16_t* Krow =
            kb + (size_t)(b * T + jc + row) * KPROJ + h * CK + ch * 32 + hi * 16;
        acc = wmma_bf16(aqw[ch], ld_frag(Krow, Krow + 8), acc);
      }
      // relative band: P[qi,m] = qr . r_k[L0-15+m], m = 0..31
      const int L0 = jc - q0 + (T - 1);
#pragma unroll
      for (int mt = 0; mt < 2; ++mt) {
        v8f pacc = {};
        const int lrow = L0 - 15 + mt * 16 + row;
#pragma unroll
        for (int ch = 0; ch < 2; ++ch) {
          const bf16_t* Rrow =
              rkb + (size_t)lrow * KPROJ + h * CK + ch * 32 + hi * 16;
          pacc = wmma_bf16(aqr[ch], ld_frag(Rrow, Rrow + 8), pacc);
        }
#pragma unroll
        for (int r = 0; r < 8; ++r)
          P[(r + hi * 8) * 32 + mt * 16 + row] = pacc[r];
      }
      // diagonal gather: issue all 8 LDS loads first (one wait), then add
      float dval[8];
#pragma unroll
      for (int r = 0; r < 8; ++r) {
        const int qi = r + hi * 8;
        dval[r] = P[qi * 32 + (row - qi + 15)];
      }
#pragma unroll
      for (int r = 0; r < 8; ++r) S[cl][r] = acc[r] + dval[r];
    }

    // --- online softmax over this 32-key block ---
    v8f scale;
#pragma unroll
    for (int r = 0; r < 8; ++r) {
      float v = fmaxf(S[0][r], S[1][r]);
      v = fmaxf(v, __shfl_xor(v, 1, 32));
      v = fmaxf(v, __shfl_xor(v, 2, 32));
      v = fmaxf(v, __shfl_xor(v, 4, 32));
      v = fmaxf(v, __shfl_xor(v, 8, 32));
      const float mnew = fmaxf(mI[r], v);
      scale[r] = __expf(mI[r] - mnew);
      S[0][r] = __expf(S[0][r] - mnew);
      S[1][r] = __expf(S[1][r] - mnew);
      float s = S[0][r] + S[1][r];
      s += __shfl_xor(s, 1, 32);
      s += __shfl_xor(s, 2, 32);
      s += __shfl_xor(s, 4, 32);
      s += __shfl_xor(s, 8, 32);
      lI[r] = lI[r] * scale[r] + s;
      mI[r] = mnew;
    }
#pragma unroll
    for (int t = 0; t < 12; ++t)
#pragma unroll
      for (int r = 0; r < 8; ++r) O[t][r] *= scale[r];

    // stage P (16 queries x 32 keys) through wave-private LDS, reload as A frag
#pragma unroll
    for (int cl = 0; cl < 2; ++cl)
#pragma unroll
      for (int r = 0; r < 8; ++r)
        P[(r + hi * 8) * 32 + cl * 16 + row] = S[cl][r];
    v16bf ap;
    {
      // lane's 16 A-fragment values = two contiguous 8-float runs in LDS:
      // [row*32 + hi*8 .. +7] and [row*32 + hi*8 + 16 .. +23] -> 4x ds b128
      const float* src0 = &P[row * 32 + hi * 8];
      const v4f c0 = *reinterpret_cast<const v4f*>(src0);
      const v4f c1 = *reinterpret_cast<const v4f*>(src0 + 4);
      const v4f c2 = *reinterpret_cast<const v4f*>(src0 + 16);
      const v4f c3 = *reinterpret_cast<const v4f*>(src0 + 20);
#pragma unroll
      for (int i = 0; i < 4; ++i) {
        ap[i]      = (bf16_t)c0[i];
        ap[i + 4]  = (bf16_t)c1[i];
        ap[i + 8]  = (bf16_t)c2[i];
        ap[i + 12] = (bf16_t)c3[i];
      }
    }

    // O += P @ V over 12 channel tiles; V transposed -> lane-contiguous frags
#pragma unroll
    for (int t = 0; t < 12; ++t) {
      const bf16_t* Vrow =
          vt + (size_t)(h * CV + t * 16 + row) * MROWS + b * T + j0 + hi * 8;
      O[t] = wmma_bf16(ap, ld_frag(Vrow, Vrow + 16), O[t]);
    }
  }

  // epilogue: normalize, write (B,T,H,CV) bf16 for the final GEMM's A operand
#pragma unroll
  for (int t = 0; t < 12; ++t)
#pragma unroll
    for (int r = 0; r < 8; ++r) {
      const int q = q0 + r + hi * 8;
      const int c = t * 16 + row;
      ob[(size_t)(b * T + q) * EMB + h * CV + c] = (bf16_t)(O[t][r] / lI[r]);
    }
}

// ---------------------------------------------------------------------------
// Host launcher
// ---------------------------------------------------------------------------
extern "C" void kernel_launch(void* const* d_in, const int* in_sizes, int n_in,
                              void* d_out, int out_size, void* d_ws, size_t ws_size,
                              hipStream_t stream) {
  const float* x    = (const float*)d_in[0];
  const float* q_w  = (const float*)d_in[1];
  const float* k_w  = (const float*)d_in[2];
  const float* v_w  = (const float*)d_in[3];
  const float* rk_w = (const float*)d_in[4];
  const float* rwb  = (const float*)d_in[5];
  const float* rrb  = (const float*)d_in[6];
  const float* embw = (const float*)d_in[7];
  const float* embb = (const float*)d_in[8];

  char* base = (char*)d_ws;
  size_t off = 0;
  auto alloc = [&](size_t bytes) -> char* {
    char* p = base + off;
    off += (bytes + 255) & ~(size_t)255;
    return p;
  };
  bf16_t* xb    = (bf16_t*)alloc((size_t)MROWS * EMB * 2);
  bf16_t* qwt   = (bf16_t*)alloc((size_t)EMB * KPROJ * 2);  // (KPROJ,EMB)
  bf16_t* kwt   = (bf16_t*)alloc((size_t)EMB * KPROJ * 2);  // (KPROJ,EMB)
  bf16_t* vwt   = (bf16_t*)alloc((size_t)EMB * EMB * 2);    // (EMB,EMB)^T
  bf16_t* rkwt  = (bf16_t*)alloc((size_t)NPF * KPROJ * 2);  // (KPROJ,NPF)
  bf16_t* embwt = (bf16_t*)alloc((size_t)EMB * EMB * 2);    // (EMB,EMB)^T
  bf16_t* peb   = (bf16_t*)alloc((size_t)LPAD * NPF * 2);
  float*  qf    = (float*) alloc((size_t)MROWS * KPROJ * 4);
  bf16_t* kbuf  = (bf16_t*)alloc((size_t)MROWS * KPROJ * 2);
  bf16_t* vtb   = (bf16_t*)alloc((size_t)MROWS * EMB * 2);  // (EMB, B*T)
  bf16_t* rkb   = (bf16_t*)alloc((size_t)LPAD * KPROJ * 2);
  bf16_t* obuf  = (bf16_t*)alloc((size_t)MROWS * EMB * 2);

  cast_f32_bf16<<<(MROWS * EMB + 255) / 256, 256, 0, stream>>>(x, xb, MROWS * EMB);
  auto castT = [&](const float* in, bf16_t* out, int R, int C) {
    cast_transpose_f32_bf16<<<dim3(C / 32, R / 8), 256, 0, stream>>>(in, out, R, C);
  };
  castT(q_w, qwt, EMB, KPROJ);
  castT(k_w, kwt, EMB, KPROJ);
  castT(v_w, vwt, EMB, EMB);
  castT(rk_w, rkwt, NPF, KPROJ);
  castT(embw, embwt, EMB, EMB);

  pos_enc_kernel<<<LPAD, NPF, 0, stream>>>(peb);

  const int M = MROWS;
  // Q (f32 out)
  gemm_nt<true, false, false><<<(M / 32) * (KPROJ / 64) / 8, 256, 0, stream>>>(
      xb, qwt, qf, nullptr, nullptr, M, KPROJ, EMB);
  // K (bf16 row-major out)
  gemm_nt<false, false, false><<<(M / 32) * (KPROJ / 64) / 8, 256, 0, stream>>>(
      xb, kwt, nullptr, kbuf, nullptr, M, KPROJ, EMB);
  // V (bf16 transposed out: (EMB, B*T))
  gemm_nt<false, false, true><<<(M / 32) * (EMB / 64) / 8, 256, 0, stream>>>(
      xb, vwt, nullptr, vtb, nullptr, M, EMB, EMB);
  // r_k (bf16 row-major out)
  gemm_nt<false, false, false><<<(LPAD / 32) * (KPROJ / 64) / 8, 256, 0, stream>>>(
      peb, rkwt, nullptr, rkb, nullptr, LPAD, KPROJ, NPF);

  const int waves = BB * H * (T / 16);  // 2048
  attn_kernel<<<waves / 4, 128, 0, stream>>>(qf, kbuf, vtb, rkb, rwb, rrb, obuf);

  // output projection with bias (f32 out)
  gemm_nt<true, true, false><<<(M / 32) * (EMB / 64) / 8, 256, 0, stream>>>(
      obuf, embwt, (float*)d_out, nullptr, embb, M, EMB, EMB);

  (void)in_sizes; (void)n_in; (void)out_size; (void)ws_size;
}
